// TemporalCrossAttention_25340307046886
// MI455X (gfx1250) — matrix operand order repaired and checked by reading, not compiled, not run on hardware
//
#include <hip/hip_runtime.h>

// Problem constants (from the reference)
#define S0c 14
#define S1c 14
#define Pc  196            // S0*S1
#define Tc  8
#define Hc  12
#define Dc  768
#define DHc 64
#define Lc  197            // P + 1
#define NTc 128
#define Nc  16             // NT / T

typedef __attribute__((ext_vector_type(16))) _Float16 v16h;
typedef __attribute__((ext_vector_type(8)))  float    v8f;

union H16 { v16h v; _Float16 h[16]; };

// A-matrix fragment (16x32 f16). Per ISA: lanes 0-15 hold row M=lane,
// K in {0..7} U {16..23}; lanes 16-31 hold row M=lane-16, K in {8..15} U {24..31}.
// rowPtr points at A[m][kk] (f32, contiguous in k). Scale folded into convert.
__device__ __forceinline__ v16h load_a_frag_f32(const float* __restrict__ rowPtr,
                                                int lane, float scale) {
  const int k0 = (lane >= 16) ? 8 : 0;
  H16 a;
#pragma unroll
  for (int j = 0; j < 8; ++j) {
    a.h[j]     = (_Float16)(rowPtr[k0 + j] * scale);
    a.h[8 + j] = (_Float16)(rowPtr[k0 + 16 + j] * scale);
  }
  return a.v;
}

// B-matrix fragment (32x16 f16): lane holds column n = lane&15;
// lanes 0-15 cover K in {0..7} U {16..23}, lanes 16-31 cover K in {8..15} U {24..31}.
// For out = A * W^T we have B[k][n] = W[n][k], so rowPtr = &W[n][kk] is
// contiguous in k.
__device__ __forceinline__ v16h load_b_frag_f32(const float* __restrict__ rowPtr,
                                                int lane) {
  const int kb = (lane < 16) ? 0 : 8;
  H16 b;
#pragma unroll
  for (int j = 0; j < 16; ++j) b.h[j] = (_Float16)rowPtr[kb + (j < 8 ? j : j + 8)];
  return b.v;
}

// ---------------------------------------------------------------------------
// Kernel 1: projection  y[m,n] = sum_k x[m,k] * W[n,k] + bias[n]
// M = NT*L = 25216 rows, N = K = 768.
// Register-blocked: one 32(M) x 64(N) tile per wave -> 2 A-frags, 4 B-frags,
// 8 WMMAs per 32-wide K step. 4 waves per block.
// ---------------------------------------------------------------------------
__global__ __launch_bounds__(128) void proj_kernel(
    const float* __restrict__ x, const float* __restrict__ W,
    const float* __restrict__ bias, float* __restrict__ y) {
  const int wid  = threadIdx.x >> 5;
  const int lane = threadIdx.x & 31;
  const int tile = blockIdx.x * 4 + wid;
  const int m0 = (tile / (Dc / 64)) * 32;   // 788 M-tiles of 32
  const int n0 = (tile % (Dc / 64)) * 64;   // 12  N-tiles of 64

  const float* arow0 = x + (size_t)(m0 + (lane & 15)) * Dc;
  const float* arow1 = arow0 + (size_t)16 * Dc;
  const float* brow  = W + (size_t)(n0 + (lane & 15)) * Dc;

  v8f c[2][4] = {};
  for (int kk = 0; kk < Dc; kk += 32) {
    // Prefetch the streaming A operand a couple of K-steps ahead
    // (lowers to global_prefetch_b8; W is L2-resident, no prefetch needed).
    __builtin_prefetch(arow0 + kk + 128, 0, 0);
    __builtin_prefetch(arow1 + kk + 128, 0, 0);

    v16h a0 = load_a_frag_f32(arow0 + kk, lane, 1.0f);
    v16h a1 = load_a_frag_f32(arow1 + kk, lane, 1.0f);
#pragma unroll
    for (int nb = 0; nb < 4; ++nb) {
      v16h b = load_b_frag_f32(brow + (size_t)nb * 16 * Dc + kk, lane);
      c[0][nb] = __builtin_amdgcn_wmma_f32_16x16x32_f16(false, a0, false, b,
                                                        (short)0, c[0][nb],
                                                        false, false);
      c[1][nb] = __builtin_amdgcn_wmma_f32_16x16x32_f16(false, a1, false, b,
                                                        (short)0, c[1][nb],
                                                        false, false);
    }
  }

  // C/D layout: VGPR i -> lanes 0-15: M=i, lanes 16-31: M=8+i; N = lane&15.
  const int rb = (lane >= 16) ? 8 : 0;
#pragma unroll
  for (int nb = 0; nb < 4; ++nb) {
    const int cidx = n0 + nb * 16 + (lane & 15);
    const float bv = bias[cidx];
#pragma unroll
    for (int mi = 0; mi < 2; ++mi) {
      const int rbase = m0 + mi * 16 + rb;
#pragma unroll
      for (int i = 0; i < 8; ++i)
        y[(size_t)(rbase + i) * Dc + cidx] = c[mi][nb][i] + bv;
    }
  }
}

// ---------------------------------------------------------------------------
// Kernel 2: attention scores + per-head softmax + head mean.
// Block = 128 threads (4 waves). grid = (13 q-tiles, 7 j, 16 n).
// half==0: q from t=j+1, k from t=j  (w1 path)
// half==1: q from t=j,   k from t=j+1 (w2 path)
// attn layout: (N*7, P, P), row index n*7+j.
// ---------------------------------------------------------------------------
__global__ __launch_bounds__(128) void attn_kernel(
    const float* __restrict__ qp, const float* __restrict__ kp,
    float* __restrict__ attn, int half) {
  __shared__ float sc[16][208];
  __shared__ float acc[16][200];

  const int qt = blockIdx.x;   // 0..12
  const int j  = blockIdx.y;   // 0..6
  const int n  = blockIdx.z;   // 0..15

  const int tq = half ? j : j + 1;
  const int tk = half ? j + 1 : j;

  const int wid  = threadIdx.x >> 5;
  const int lane = threadIdx.x & 31;

  const float* qbase = qp + ((size_t)(n * Tc + tq) * Lc + 1) * Dc;
  const float* kbase = kp + ((size_t)(n * Tc + tk) * Lc + 1) * Dc;

  for (int i = threadIdx.x; i < 16 * Pc; i += blockDim.x)
    acc[i / Pc][i % Pc] = 0.0f;
  __syncthreads();

  const int q0   = qt * 16;
  const int qrow = min(q0 + (lane & 15), Pc - 1);

  for (int h = 0; h < Hc; ++h) {
    const float* aptr = qbase + (size_t)qrow * Dc + h * DHc;
    // Q fragments depend only on h: load once, reuse across all K-tiles.
    const v16h aLo = load_a_frag_f32(aptr,      lane, 0.125f);  // 1/sqrt(64)
    const v16h aHi = load_a_frag_f32(aptr + 32, lane, 0.125f);

    for (int nt = wid; nt < 13; nt += 4) {
      const int krow = min(nt * 16 + (lane & 15), Pc - 1);
      const float* bptr = kbase + (size_t)krow * Dc + h * DHc;
      v16h b0 = load_b_frag_f32(bptr,      lane);
      v16h b1 = load_b_frag_f32(bptr + 32, lane);
      v8f c = {};
      c = __builtin_amdgcn_wmma_f32_16x16x32_f16(false, aLo, false, b0,
                                                 (short)0, c, false, false);
      c = __builtin_amdgcn_wmma_f32_16x16x32_f16(false, aHi, false, b1,
                                                 (short)0, c, false, false);
      const int rbw = (lane >= 16) ? 8 : 0;
      const int cc  = nt * 16 + (lane & 15);
#pragma unroll
      for (int i = 0; i < 8; ++i) sc[rbw + i][cc] = c[i];
    }
    __syncthreads();

    // Row-wise softmax over cols [0,196); 8 threads cooperate per row.
    {
      const int r   = threadIdx.x >> 3;  // 0..15
      const int sub = threadIdx.x & 7;
      float mx = -3.4e38f;
      for (int c0 = sub; c0 < Pc; c0 += 8) mx = fmaxf(mx, sc[r][c0]);
#pragma unroll
      for (int off = 4; off > 0; off >>= 1) mx = fmaxf(mx, __shfl_xor(mx, off, 32));
      float sum = 0.0f;
      for (int c0 = sub; c0 < Pc; c0 += 8) {
        float e = __expf(sc[r][c0] - mx);
        sc[r][c0] = e;
        sum += e;
      }
#pragma unroll
      for (int off = 4; off > 0; off >>= 1) sum += __shfl_xor(sum, off, 32);
      const float inv = 1.0f / sum;
      for (int c0 = sub; c0 < Pc; c0 += 8) acc[r][c0] += sc[r][c0] * inv;
    }
    __syncthreads();
  }

  // Write mean over heads.
  const int r    = threadIdx.x >> 3;
  const int sub  = threadIdx.x & 7;
  const int grow = q0 + r;
  if (grow < Pc) {
    float* dst = attn + ((size_t)(n * 7 + j) * Pc + grow) * Pc;
    for (int c0 = sub; c0 < Pc; c0 += 8) dst[c0] = acc[r][c0] * (1.0f / 12.0f);
  }
}

// ---------------------------------------------------------------------------
// Kernel 3: out[(n,t,1+qq), c] += sum_k attn[(n,j), qq, k] * w[idx(qq,k), c]
// For fixed qq: GEMM M=112 (n*7+j), K=196 (padded to 224), N=768.
// One wave owns (qq, 16-wide N-tile) and loops all 7 M-tiles, so the
// expensive gathered B fragment is built once per K-step and reused 7x.
// Launched once per half (stream-serialized so += on d_out is race-free).
// ---------------------------------------------------------------------------
__global__ __launch_bounds__(128) void outacc_kernel(
    const float* __restrict__ attn, const float* __restrict__ w,
    float* __restrict__ out, int half) {
  const int wid  = threadIdx.x >> 5;
  const int lane = threadIdx.x & 31;
  const int tile = blockIdx.x * 4 + wid;    // 196*48 = 9408 tiles
  const int qq = tile / 48;
  const int nt = tile % 48;

  const int qi = qq / S1c, qj = qq % S1c;
  const int cidx = nt * 16 + (lane & 15);

  v8f c[7] = {};
  for (int kk = 0; kk < 224; kk += 32) {
    // B fragment: rows of w gathered via the relative-position index;
    // built once per K-step, shared by all 7 M-tiles.
    const int kbb = kk + ((lane < 16) ? 0 : 8);
    H16 b;
#pragma unroll
    for (int jj = 0; jj < 16; ++jj) {
      const int k = kbb + (jj < 8 ? jj : jj + 8);
      _Float16 v = (_Float16)0.0f;
      if (k < Pc) {
        const int ki = k / S1c, kj = k % S1c;
        const int idx = (qi - ki + S0c - 1) * (2 * S1c - 1) + (qj - kj + S1c - 1);
        v = (_Float16)w[(size_t)idx * Dc + cidx];
      }
      b.h[jj] = v;
    }

    const int k0 = kk + ((lane >= 16) ? 8 : 0);
#pragma unroll
    for (int mt = 0; mt < 7; ++mt) {
      const float* arow =
          attn + ((size_t)(mt * 16 + (lane & 15)) * Pc + qq) * Pc;
      H16 a;  // A fragment, zero-padded for k >= 196
#pragma unroll
      for (int jj = 0; jj < 8; ++jj) {
        const int ka = k0 + jj, kb2 = k0 + 16 + jj;
        a.h[jj]     = (ka  < Pc) ? (_Float16)arow[ka]  : (_Float16)0.0f;
        a.h[8 + jj] = (kb2 < Pc) ? (_Float16)arow[kb2] : (_Float16)0.0f;
      }
      c[mt] = __builtin_amdgcn_wmma_f32_16x16x32_f16(false, a.v, false, b.v,
                                                     (short)0, c[mt],
                                                     false, false);
    }
  }

  const int rb = (lane >= 16) ? 8 : 0;
#pragma unroll
  for (int mt = 0; mt < 7; ++mt) {
#pragma unroll
    for (int i = 0; i < 8; ++i) {
      const int mr = mt * 16 + rb + i;   // 0..111 == n*7+j
      const int nn = mr / 7;
      const int jj = mr % 7;
      const int t  = half ? jj : jj + 1;
      float* dst = out + ((size_t)(nn * Tc + t) * Lc + 1 + qq) * Dc + cidx;
      *dst += c[mt][i];
    }
  }
}

// ---------------------------------------------------------------------------
extern "C" void kernel_launch(void* const* d_in, const int* in_sizes, int n_in,
                              void* d_out, int out_size, void* d_ws, size_t ws_size,
                              hipStream_t stream) {
  (void)in_sizes; (void)n_in; (void)ws_size;
  const float* q  = (const float*)d_in[0];
  const float* k  = (const float*)d_in[1];
  const float* Wq = (const float*)d_in[2];
  const float* bq = (const float*)d_in[3];
  const float* Wk = (const float*)d_in[4];
  const float* bk = (const float*)d_in[5];
  const float* w1 = (const float*)d_in[6];
  const float* w2 = (const float*)d_in[7];
  float* out = (float*)d_out;

  const size_t projElems = (size_t)NTc * Lc * Dc;        // 25216*768
  const size_t attnElems = (size_t)Nc * 7 * Pc * Pc;     // 16*7*196*196
  float* qp = (float*)d_ws;
  float* kp = qp + projElems;
  float* a1 = kp + projElems;
  float* a2 = a1 + attnElems;

  hipMemsetAsync(out, 0, (size_t)out_size * sizeof(float), stream);

  // Projections: tiles = (25216/32) * (768/64) = 788*12 = 9456; 4 waves/block.
  const int projBlocks = ((NTc * Lc / 32) * (Dc / 64)) / 4;   // 2364
  proj_kernel<<<projBlocks, 128, 0, stream>>>(q, Wq, bq, qp);
  proj_kernel<<<projBlocks, 128, 0, stream>>>(k, Wk, bk, kp);

  // Attention + softmax + head-mean.
  dim3 ga(13, 7, Nc);
  attn_kernel<<<ga, 128, 0, stream>>>(qp, kp, a1, 0);
  attn_kernel<<<ga, 128, 0, stream>>>(qp, kp, a2, 1);

  // Output accumulation: tiles = 196 * 48 = 9408; 4 waves/block.
  const int outBlocks = (Pc * 48) / 4;                        // 2352
  outacc_kernel<<<outBlocks, 128, 0, stream>>>(a1, w1, out, 0);
  outacc_kernel<<<outBlocks, 128, 0, stream>>>(a2, w2, out, 1);
}
